// GNN_prelu_edge_32822140076340
// MI455X (gfx1250) — matrix operand order repaired and checked by reading, not compiled, not run on hardware
//
#include <hip/hip_runtime.h>

typedef __attribute__((ext_vector_type(16))) _Float16 v16h;
typedef __attribute__((ext_vector_type(8)))  float    v8f;

#define DOUT 64
#define BN_EPS 1e-5f

// ---------------------------------------------------------------------------
// Edge scatter: agg[col] += xsrc[row], cnt[col] += 1   (for segment-mean)
// ---------------------------------------------------------------------------
__global__ void k_scatter(const float* __restrict__ xsrc,
                          const int* __restrict__ row, const int* __restrict__ col,
                          float* __restrict__ agg, float* __restrict__ cnt,
                          int E, int din) {
  int gid = blockIdx.x * blockDim.x + threadIdx.x;
  if (gid >= E * din) return;
  int e = gid / din;
  int f = gid - e * din;
  int c = col[e];
  atomicAdd(&agg[c * din + f], xsrc[row[e] * din + f]);
  if (f == 0) atomicAdd(&cnt[c], 1.0f);
}

// ---------------------------------------------------------------------------
// Node linear: pre[i,f] = (agg[i]/max(cnt,1)) . Wl[:,f] + bl[f] + xdst[i] . Wr[:,f]
// ---------------------------------------------------------------------------
__global__ void k_node_linear(const float* __restrict__ agg, const float* __restrict__ cnt,
                              const float* __restrict__ xdst,
                              const float* __restrict__ Wl, const float* __restrict__ bl,
                              const float* __restrict__ Wr,
                              float* __restrict__ pre, int n, int din) {
  int gid = blockIdx.x * blockDim.x + threadIdx.x;
  if (gid >= n * DOUT) return;
  int i = gid >> 6, f = gid & 63;
  float cinv = 1.0f / fmaxf(cnt[i], 1.0f);
  float s = bl[f];
  for (int k = 0; k < din; ++k)
    s += (agg[i * din + k] * cinv) * Wl[k * DOUT + f] + xdst[i * din + k] * Wr[k * DOUT + f];
  pre[gid] = s;
}

// Load 8 contiguous floats, convert to f16 into A fragment elements [off..off+7]
__device__ __forceinline__ void loadA8(const float* p, v16h& a, int off) {
  float4 u = *(const float4*)p;
  float4 v = *(const float4*)(p + 4);
  a[off + 0] = (_Float16)u.x; a[off + 1] = (_Float16)u.y;
  a[off + 2] = (_Float16)u.z; a[off + 3] = (_Float16)u.w;
  a[off + 4] = (_Float16)v.x; a[off + 5] = (_Float16)v.y;
  a[off + 6] = (_Float16)v.z; a[off + 7] = (_Float16)v.w;
}

// ---------------------------------------------------------------------------
// Gate kernel (WMMA hotspot). One wave = 16 edges.
//   ea      = ea_raw @ We + be                (recomputed into LDS, f32)
//   logits  = [pre[col] | ea] @ Wg + bg      (v_wmma_f32_16x16x32_f16)
//   post[col] += sigmoid(logits) * ea        (global f32 atomics)
// ---------------------------------------------------------------------------
__global__ void k_gate_wmma(const float* __restrict__ pre, float* __restrict__ post,
                            const int* __restrict__ col,
                            const float* __restrict__ ea_raw,
                            const float* __restrict__ We, const float* __restrict__ be,
                            const float* __restrict__ Wg, const float* __restrict__ bg,
                            int E) {
  // B operand in fragment-major layout: sWgT[ncol*128 + k] = Wg[k, ncol] (f16).
  // A lane's B fragment = 16 contiguous halves -> one aligned v16h (2x ds_load_b128).
  __shared__ __align__(32) _Float16 sWgT[DOUT * 128];  // 16 KB
  __shared__ __align__(16) float    sEa[8][16 * DOUT]; // 32 KB per-wave edge attrs
  __shared__ float sWe[8 * DOUT];                      // 2 KB
  __shared__ float sBe[DOUT];
  __shared__ int   sCol[8][16];

  const int tid  = threadIdx.x;
  const int lane = tid & 31;
  const int w    = tid >> 5;

  // Stage Wg (transposed, f16), We, be into LDS — coalesced global reads
  for (int idx = tid; idx < 128 * DOUT; idx += 256) {
    int k = idx >> 6, ncol = idx & 63;
    sWgT[ncol * 128 + k] = (_Float16)Wg[idx];
  }
  for (int idx = tid; idx < 8 * DOUT; idx += 256) sWe[idx] = We[idx];
  if (tid < DOUT) sBe[tid] = be[tid];
  __syncthreads();

  const int e0 = (blockIdx.x * 8 + w) * 16;

  // ea: 2 lanes per edge, each produces 32 contiguous output features
  {
    int m   = lane >> 1;
    int kb0 = (lane & 1) * 32;
    int e = e0 + m; if (e >= E) e = E - 1;
    float4 a0 = *(const float4*)(ea_raw + (size_t)e * 8);
    float4 a1 = *(const float4*)(ea_raw + (size_t)e * 8 + 4);
    float er[8] = {a0.x, a0.y, a0.z, a0.w, a1.x, a1.y, a1.z, a1.w};
    for (int k = 0; k < 32; ++k) {
      int kc = kb0 + k;
      float v = sBe[kc];
#pragma unroll
      for (int j = 0; j < 8; ++j) v += er[j] * sWe[j * DOUT + kc];
      sEa[w][m * DOUT + kc] = v;
    }
  }
  if (lane < 16) { int e = e0 + lane; if (e >= E) e = E - 1; sCol[w][lane] = col[e]; }
  __syncthreads();

  if (e0 >= E) return;  // wave-uniform: EXEC stays all-ones through the WMMAs

  const int half = lane >> 4;      // lane half-group
  const int r    = lane & 15;      // A-matrix row (edge within tile) for this lane
  const int n    = lane & 15;      // B/D column for this lane
  const int c    = sCol[w][r];
  const float* preRow = pre + (size_t)c * DOUT;
  const float* eaRow  = &sEa[w][r * DOUT];

  // A fragments in contiguous K order (matches B):
  //   elems 0..15 -> K = kb*32 + half*16 + j
  v16h A[4];
#pragma unroll
  for (int kb = 0; kb < 2; ++kb) {        // K 0..63: gathered pre[col]
    int bk = kb * 32 + half * 16;
    loadA8(preRow + bk,      A[kb], 0);
    loadA8(preRow + bk + 8, A[kb], 8);
  }
#pragma unroll
  for (int kb = 2; kb < 4; ++kb) {        // K 64..127: ea from LDS
    int bk = kb * 32 + half * 16 - 64;
    loadA8(eaRow + bk,      A[kb], 0);
    loadA8(eaRow + bk + 8, A[kb], 8);
  }

#pragma unroll
  for (int nt = 0; nt < 4; ++nt) {
    v8f acc = {};
    const _Float16* bcol = sWgT + (nt * 16 + n) * 128 + half * 16;
#pragma unroll
    for (int kb = 0; kb < 4; ++kb) {
      v16h B = *(const v16h*)(bcol + kb * 32);   // 16 contiguous halves, 32B aligned
      acc = __builtin_amdgcn_wmma_f32_16x16x32_f16(false, A[kb], false, B,
                                                   (short)0, acc, false, false);
    }
    // D layout: lane n = column, VGPR rr = row rr (lanes 0-15) / rr+8 (lanes 16-31)
    int   ncol = nt * 16 + n;
    float bgv  = bg[ncol];
#pragma unroll
    for (int rr = 0; rr < 8; ++rr) {
      int m = half * 8 + rr;
      if (e0 + m < E) {
        float g   = 1.0f / (1.0f + __expf(-(acc[rr] + bgv)));
        float val = g * sEa[w][m * DOUT + ncol];
        atomicAdd(&post[sCol[w][m] * DOUT + ncol], val);
      }
    }
  }
}

// ---------------------------------------------------------------------------
// BatchNorm statistics: stats[f] = sum, stats[64+f] = sumsq
// ---------------------------------------------------------------------------
__global__ void k_bn_stats(const float* __restrict__ post, float* __restrict__ stats, int n) {
  int gid    = blockIdx.x * blockDim.x + threadIdx.x;
  int f      = gid & 63;
  int i0     = gid >> 6;
  int stride = (gridDim.x * blockDim.x) >> 6;
  float s = 0.0f, ss = 0.0f;
  for (int i = i0; i < n; i += stride) {
    float v = post[i * DOUT + f];
    s += v; ss += v * v;
  }
  atomicAdd(&stats[f], s);
  atomicAdd(&stats[DOUT + f], ss);
}

// BN (biased var) + affine + relu(2x), accumulate into HeteroConv sum buffer
__global__ void k_bn_relu_accum(const float* __restrict__ post, const float* __restrict__ stats,
                                const float* __restrict__ gamma, const float* __restrict__ beta,
                                float* __restrict__ acc, int n) {
  int gid = blockIdx.x * blockDim.x + threadIdx.x;
  if (gid >= n * DOUT) return;
  int f = gid & 63;
  float inv = 1.0f / (float)n;
  float mu  = stats[f] * inv;
  float var = stats[DOUT + f] * inv - mu * mu;
  float o = (post[gid] - mu) * rsqrtf(var + BN_EPS) * gamma[f] + beta[f];
  acc[gid] += fmaxf(2.0f * o, 0.0f);
}

// Plain SAGE over identity self-loop edges: agg == x, cnt == 1
__global__ void k_plain_sage_accum(const float* __restrict__ x,
                                   const float* __restrict__ Wl, const float* __restrict__ bl,
                                   const float* __restrict__ Wr,
                                   float* __restrict__ acc, int n, int din) {
  int gid = blockIdx.x * blockDim.x + threadIdx.x;
  if (gid >= n * DOUT) return;
  int i = gid >> 6, f = gid & 63;
  float s = bl[f];
  for (int k = 0; k < din; ++k)
    s += x[i * din + k] * (Wl[k * DOUT + f] + Wr[k * DOUT + f]);
  acc[gid] += s;
}

__global__ void k_relu(const float* __restrict__ acc, float* __restrict__ h, int tot) {
  int gid = blockIdx.x * blockDim.x + threadIdx.x;
  if (gid < tot) h[gid] = fmaxf(acc[gid], 0.0f);
}

// PReLU (identity on post-relu data, kept faithful) + linear head to scalar
__global__ void k_head(const float* __restrict__ h, const float* __restrict__ a,
                       const float* __restrict__ W, const float* __restrict__ b,
                       float* __restrict__ out, int n) {
  int i = blockIdx.x * blockDim.x + threadIdx.x;
  if (i >= n) return;
  float s = b[0];
  for (int k = 0; k < DOUT; ++k) {
    float v = h[i * DOUT + k];
    v = (v >= 0.0f) ? v : a[k] * v;
    s += v * W[k];
  }
  out[i] = s;
}

// ---------------------------------------------------------------------------
// Host orchestration
// ---------------------------------------------------------------------------
struct GPar { const float *Wl,*bl,*Wr,*We,*be,*Wg,*bg,*gamma,*beta; };
struct PPar { const float *Wl,*bl,*Wr; };

extern "C" void kernel_launch(void* const* d_in, const int* in_sizes, int n_in,
                              void* d_out, int out_size, void* d_ws, size_t ws_size,
                              hipStream_t stream) {
  const int DIN = 16;
  const float* xP = (const float*)d_in[0];
  const float* xG = (const float*)d_in[1];
  const float* xS = (const float*)d_in[2];
  const int nP = in_sizes[0] / DIN;
  const int nG = in_sizes[1] / DIN;
  const int nS = in_sizes[2] / DIN;

  const int* ei_pg = (const int*)d_in[3]; const int Epg = in_sizes[3] / 2;
  const int* ei_gp = (const int*)d_in[4]; const int Egp = in_sizes[4] / 2;
  const int* ei_ps = (const int*)d_in[5]; const int Eps = in_sizes[5] / 2;
  const int* ei_sp = (const int*)d_in[6]; const int Esp = in_sizes[6] / 2;
  const int* ei_gg = (const int*)d_in[7]; const int Egg = in_sizes[7] / 2;
  const float* ea_pg = (const float*)d_in[9];
  const float* ea_gp = (const float*)d_in[10];
  const float* ea_ps = (const float*)d_in[11];
  const float* ea_sp = (const float*)d_in[12];
  const float* ea_gg = (const float*)d_in[13];

  int ip = 14;
  auto nextG = [&]() {
    GPar g;
    g.Wl=(const float*)d_in[ip++]; g.bl=(const float*)d_in[ip++]; g.Wr=(const float*)d_in[ip++];
    g.We=(const float*)d_in[ip++]; g.be=(const float*)d_in[ip++];
    g.Wg=(const float*)d_in[ip++]; g.bg=(const float*)d_in[ip++];
    g.gamma=(const float*)d_in[ip++]; g.beta=(const float*)d_in[ip++];
    return g;
  };
  auto nextP = [&]() {
    PPar p;
    p.Wl=(const float*)d_in[ip++]; p.bl=(const float*)d_in[ip++]; p.Wr=(const float*)d_in[ip++];
    return p;
  };
  GPar pg1=nextG(), gp1=nextG(), ps1=nextG(), sp1=nextG(), gg1=nextG(); PPar sl1=nextP();
  GPar pg2=nextG(), gp2=nextG(), ps2=nextG(), sp2=nextG(), gg2=nextG(); PPar sl2=nextP();
  const float* a_gw   = (const float*)d_in[ip++];
  const float* a_sw   = (const float*)d_in[ip++];
  const float* Wout_g = (const float*)d_in[ip++];
  const float* bout_g = (const float*)d_in[ip++];
  const float* Wout_s = (const float*)d_in[ip++];
  const float* bout_s = (const float*)d_in[ip++];

  // Workspace carve (floats)
  float* ws = (float*)d_ws;
  size_t off = 0;
  auto carve = [&](size_t nf) { float* p = ws + off; off += nf; return p; };
  float* hP   = carve((size_t)nP * DOUT);
  float* hG   = carve((size_t)nG * DOUT);
  float* hS   = carve((size_t)nS * DOUT);
  float* accP = carve((size_t)nP * DOUT);
  float* accG = carve((size_t)nG * DOUT);
  float* accS = carve((size_t)nS * DOUT);
  float* pre  = carve((size_t)nG * DOUT);
  float* post = carve((size_t)nG * DOUT);
  float* agg  = carve((size_t)nG * DOUT);   // max(n_dst*din)
  float* cnt  = carve((size_t)nG);
  float* stats= carve(2 * DOUT);

  auto gated = [&](const float* xsrc, const float* xdst, const int* ei, int E,
                   const float* ear, const GPar& p, float* acc, int n_dst, int din) {
    hipMemsetAsync(agg, 0, (size_t)n_dst * din * sizeof(float), stream);
    hipMemsetAsync(cnt, 0, (size_t)n_dst * sizeof(float), stream);
    int totE = E * din;
    k_scatter<<<(totE + 255) / 256, 256, 0, stream>>>(xsrc, ei, ei + E, agg, cnt, E, din);
    int totN = n_dst * DOUT;
    k_node_linear<<<(totN + 255) / 256, 256, 0, stream>>>(agg, cnt, xdst, p.Wl, p.bl, p.Wr,
                                                          pre, n_dst, din);
    hipMemcpyAsync(post, pre, (size_t)totN * sizeof(float), hipMemcpyDeviceToDevice, stream);
    k_gate_wmma<<<(E + 127) / 128, 256, 0, stream>>>(pre, post, ei + E, ear,
                                                     p.We, p.be, p.Wg, p.bg, E);
    hipMemsetAsync(stats, 0, 2 * DOUT * sizeof(float), stream);
    k_bn_stats<<<64, 256, 0, stream>>>(post, stats, n_dst);
    k_bn_relu_accum<<<(totN + 255) / 256, 256, 0, stream>>>(post, stats, p.gamma, p.beta,
                                                            acc, n_dst);
  };

  auto run_layer = [&](const float* inP, const float* inG, const float* inS, int din,
                       const GPar& pg, const GPar& gp, const GPar& ps, const GPar& sp,
                       const GPar& gg, const PPar& sl) {
    hipMemsetAsync(accP, 0, (size_t)nP * DOUT * sizeof(float), stream);
    hipMemsetAsync(accG, 0, (size_t)nG * DOUT * sizeof(float), stream);
    hipMemsetAsync(accS, 0, (size_t)nS * DOUT * sizeof(float), stream);
    // dst = gw
    gated(inP, inG, ei_pg, Epg, ea_pg, pg, accG, nG, din);
    gated(inG, inG, ei_gg, Egg, ea_gg, gg, accG, nG, din);
    k_plain_sage_accum<<<(nG * DOUT + 255) / 256, 256, 0, stream>>>(inG, sl.Wl, sl.bl, sl.Wr,
                                                                    accG, nG, din);
    // dst = pfas
    gated(inG, inP, ei_gp, Egp, ea_gp, gp, accP, nP, din);
    gated(inS, inP, ei_sp, Esp, ea_sp, sp, accP, nP, din);
    // dst = sw
    gated(inP, inS, ei_ps, Eps, ea_ps, ps, accS, nS, din);
    // final relu of HeteroConv sums
    k_relu<<<(nP * DOUT + 255) / 256, 256, 0, stream>>>(accP, hP, nP * DOUT);
    k_relu<<<(nG * DOUT + 255) / 256, 256, 0, stream>>>(accG, hG, nG * DOUT);
    k_relu<<<(nS * DOUT + 255) / 256, 256, 0, stream>>>(accS, hS, nS * DOUT);
  };

  run_layer(xP, xG, xS, 16, pg1, gp1, ps1, sp1, gg1, sl1);
  run_layer(hP, hG, hS, 64, pg2, gp2, ps2, sp2, gg2, sl2);

  // Heads + output packing: [gw (nG), sw (nS), xp (nP*64)]
  float* outF = (float*)d_out;
  k_head<<<(nG + 255) / 256, 256, 0, stream>>>(hG, a_gw, Wout_g, bout_g, outF, nG);
  k_head<<<(nS + 255) / 256, 256, 0, stream>>>(hS, a_sw, Wout_s, bout_s, outF + nG, nS);
  hipMemcpyAsync(outF + nG + nS, hP, (size_t)nP * DOUT * sizeof(float),
                 hipMemcpyDeviceToDevice, stream);
  (void)n_in; (void)out_size; (void)ws_size;
}
